// GroupSupConLoss_13091060319094
// MI455X (gfx1250) — matrix operand order repaired and checked by reading, not compiled, not run on hardware
//
#include <hip/hip_runtime.h>
#include <hip/hip_bf16.h>

typedef __attribute__((ext_vector_type(16))) _Float16 v16h;
typedef __attribute__((ext_vector_type(8)))  _Float16 h8;
typedef __attribute__((ext_vector_type(4)))  _Float16 h4;
typedef __attribute__((ext_vector_type(8)))  float    v8f;
typedef __attribute__((ext_vector_type(4)))  int      v4i;

#define NROWS 8192      // 2B
#define DDIM  128
#define E2    7.3890560989306495f   // exp(2.0) = exp(sim[r,r]/T), removed from row sums
#define CC    64        // columns staged in LDS per chunk
#define LDST  136       // padded LDS row stride in halves (272B -> 4-bank rotation per row)
#define NCHUNK (1024 / CC)

#if __has_builtin(__builtin_amdgcn_global_load_async_to_lds_b128) && \
    __has_builtin(__builtin_amdgcn_s_wait_asynccnt)
#define HAVE_ASYNC 1
#else
#define HAVE_ASYNC 0
#endif

// Build a v16h fragment from two 16-byte chunks.
__device__ inline v16h ld2x8(const _Float16* p0, const _Float16* p1) {
  union { v16h v; h8 h[2]; } u;
  u.h[0] = *(const h8*)p0;
  u.h[1] = *(const h8*)p1;
  return u.v;
}

// ---------------- Kernel 1: row-normalize, split into f16 hi+lo, zero accumulators
__global__ void k_normalize(const float* __restrict__ ei, const float* __restrict__ ej,
                            _Float16* __restrict__ zhi, _Float16* __restrict__ zlo,
                            float* __restrict__ denom, float* __restrict__ accum) {
  const int lane = threadIdx.x & 31;
  const int wave = threadIdx.x >> 5;
  const int row  = blockIdx.x * 8 + wave;           // 1024 blocks * 8 waves = 8192 rows
  const int g    = blockIdx.x * 256 + threadIdx.x;
  if (g < NROWS) denom[g] = 0.0f;
  if (g == 0)    *accum   = 0.0f;

  const float* src = (row < 4096) ? (ei + (size_t)row * DDIM)
                                  : (ej + (size_t)(row - 4096) * DDIM);
  float4 x = ((const float4*)src)[lane];            // 32 lanes * 4 = 128
  float ss = x.x*x.x + x.y*x.y + x.z*x.z + x.w*x.w;
#pragma unroll
  for (int m = 16; m >= 1; m >>= 1) ss += __shfl_xor(ss, m, 32);
  const float inv = 1.0f / fmaxf(sqrtf(ss), 1e-12f);

  float zf[4] = { x.x*inv, x.y*inv, x.z*inv, x.w*inv };
  h4 hi, lo;
#pragma unroll
  for (int i = 0; i < 4; ++i) {
    _Float16 h = (_Float16)zf[i];
    hi[i] = h;
    lo[i] = (_Float16)(zf[i] - (float)h);
  }
  *(h4*)(zhi + (size_t)row * DDIM + lane * 4) = hi;
  *(h4*)(zlo + (size_t)row * DDIM + lane * 4) = lo;
}

// ---------------- Kernel 2: positive-pair terms:  accum += sum_b  -4 * cos(emb_i[b], emb_j[b])
// (each pair appears twice in the 2B rows, and 1/T = 2  =>  factor 4)
__global__ void k_pos(const float* __restrict__ ei, const float* __restrict__ ej,
                      float* __restrict__ accum) {
  __shared__ float blk;
  if (threadIdx.x == 0) blk = 0.0f;
  __syncthreads();

  const int lane = threadIdx.x & 31;
  const int wave = threadIdx.x >> 5;
  const int b    = blockIdx.x * 8 + wave;           // 512 blocks * 8 waves = 4096 pairs
  float4 xi = ((const float4*)(ei + (size_t)b * DDIM))[lane];
  float4 xj = ((const float4*)(ej + (size_t)b * DDIM))[lane];
  float ii = xi.x*xi.x + xi.y*xi.y + xi.z*xi.z + xi.w*xi.w;
  float jj = xj.x*xj.x + xj.y*xj.y + xj.z*xj.z + xj.w*xj.w;
  float ij = xi.x*xj.x + xi.y*xj.y + xi.z*xj.z + xi.w*xj.w;
#pragma unroll
  for (int m = 16; m >= 1; m >>= 1) {
    ii += __shfl_xor(ii, m, 32);
    jj += __shfl_xor(jj, m, 32);
    ij += __shfl_xor(ij, m, 32);
  }
  if (lane == 0) {
    float p = ij / (fmaxf(sqrtf(ii), 1e-12f) * fmaxf(sqrtf(jj), 1e-12f));
    atomicAdd(&blk, -4.0f * p);
  }
  __syncthreads();
  if (threadIdx.x == 0) atomicAdd(accum, blk);
}

// ---------------- Kernel 3: fused sim-GEMM + exp + row-sum (the heavy one)
// grid = (64 row-blocks, 8 column-slices); block = 256 = 8 waves; wave owns a 16-row tile.
// sim = (zhi+zlo)(zhi+zlo)^T  ~=  hi*hi + hi*lo + lo*hi   (3 WMMAs / K-tile, ~22-bit mantissa)
// Column chunks double-buffered through LDS via GLOBAL_LOAD_ASYNC_TO_LDS_B128 (ASYNCcnt pipeline).
__global__ void __launch_bounds__(256) k_sim(const _Float16* __restrict__ zhi,
                                             const _Float16* __restrict__ zlo,
                                             float* __restrict__ denom) {
  __shared__ _Float16 sh[2][2][CC][LDST];           // [buf][hi/lo][col][padded K] ~68 KB

  const int lane   = threadIdx.x & 31;
  const int wave   = threadIdx.x >> 5;
  const int hs     = lane >> 4;                     // half-wave select (K interleave)
  const int m_base = blockIdx.x * 128 + wave * 16;
  const int col0   = blockIdx.y * 1024;             // 8 slices * 1024 = full 8192 columns

  // Load the 8 A fragments once; resident in VGPRs for the whole column sweep.
  // ISA 16-bit A 16x32 layout: lane(l) holds row m=l%16, halves [kt*32+hs*8,+8) and [kt*32+16+hs*8,+8)
  const size_t arow = (size_t)(m_base + (lane & 15)) * DDIM;
  v16h a_hi[4], a_lo[4];
#pragma unroll
  for (int kt = 0; kt < 4; ++kt) {
    const _Float16* ph = zhi + arow + kt * 32 + hs * 8;
    const _Float16* pl = zlo + arow + kt * 32 + hs * 8;
    a_hi[kt] = ld2x8(ph, ph + 16);
    a_lo[kt] = ld2x8(pl, pl + 16);
  }

  // Per-thread staging role: arr = hi/lo stream, cpart = 16B piece, rows r0, r0+8, ..., r0+56.
  const int arr   = threadIdx.x & 1;
  const int cpart = (threadIdx.x >> 1) & 15;
  const int r0    = threadIdx.x >> 5;
  const _Float16* zb = arr ? zlo : zhi;

  auto issue_chunk = [&](int buf, int cbase) {
#pragma unroll
    for (int k = 0; k < 8; ++k) {
      const int r = r0 + 8 * k;
      const _Float16* src = zb + (size_t)(col0 + cbase + r) * DDIM + cpart * 8;
      _Float16* dst = &sh[buf][arr][r][cpart * 8];
#if HAVE_ASYNC
      __builtin_amdgcn_global_load_async_to_lds_b128(
          (__attribute__((address_space(1))) v4i*)src,
          (__attribute__((address_space(3))) v4i*)dst, 0, 0);
#else
      *(float4*)dst = *(const float4*)src;
#endif
    }
  };

  float rs[8];
#pragma unroll
  for (int i = 0; i < 8; ++i) rs[i] = 0.0f;

  issue_chunk(0, 0);                                // prologue: fill buffer 0
  for (int cidx = 0; cidx < NCHUNK; ++cidx) {
    const int buf = cidx & 1;
    if (cidx + 1 < NCHUNK) issue_chunk(buf ^ 1, (cidx + 1) * CC);
#if HAVE_ASYNC
    // 8 async ops per wave per chunk; loads complete in order, so <=8 outstanding
    // guarantees the current buffer is fully in LDS while the next streams in.
    if (cidx + 1 < NCHUNK) __builtin_amdgcn_s_wait_asynccnt(8);
    else                   __builtin_amdgcn_s_wait_asynccnt(0);
#endif
    __syncthreads();

#pragma unroll
    for (int ct = 0; ct < CC / 16; ++ct) {
      v8f c = {};
      const int coln = ct * 16 + (lane & 15);
#pragma unroll
      for (int kt = 0; kt < 4; ++kt) {
        // ISA 16-bit B 32x16 layout: lane(l) holds col n=l%16, halves [kt*32+hs*8,+8) and [kt*32+16+hs*8,+8)
        const _Float16* bh = &sh[buf][0][coln][kt * 32 + hs * 8];
        const _Float16* bl = &sh[buf][1][coln][kt * 32 + hs * 8];
        v16h bhi = ld2x8(bh, bh + 16);
        v16h blo = ld2x8(bl, bl + 16);
        c = __builtin_amdgcn_wmma_f32_16x16x32_f16(false, a_hi[kt], false, bhi, (short)0, c, false, false);
        c = __builtin_amdgcn_wmma_f32_16x16x32_f16(false, a_hi[kt], false, blo, (short)0, c, false, false);
        c = __builtin_amdgcn_wmma_f32_16x16x32_f16(false, a_lo[kt], false, bhi, (short)0, c, false, false);
      }
      // epilogue: exp(sim/T) = exp(2*sim), accumulate per-lane row partials (defer lane reduction)
#pragma unroll
      for (int i = 0; i < 8; ++i) rs[i] += __expf(2.0f * c[i]);
    }
    __syncthreads();                                // compute done before buffer reuse
  }

  // One 16-lane reduction per row at the very end (C layout: VGPR i -> row i (+8 for upper half-wave))
#pragma unroll
  for (int i = 0; i < 8; ++i) {
    float v = rs[i];
#pragma unroll
    for (int m = 8; m >= 1; m >>= 1) v += __shfl_xor(v, m, 16);
    rs[i] = v;
  }
  if ((lane & 15) == 0) {
#pragma unroll
    for (int i = 0; i < 8; ++i)
      atomicAdd(&denom[m_base + i + 8 * hs], rs[i]);
  }
}

// ---------------- Kernel 4: loss = (accum + sum_r log(denom_r - e^2)) / 2B
__global__ void k_final(const float* __restrict__ denom, const float* __restrict__ accum,
                        float* __restrict__ out) {
  __shared__ float red[256];
  float s = 0.0f;
  for (int r = threadIdx.x; r < NROWS; r += 256)
    s += __logf(fmaxf(denom[r] - E2, 1e-30f));
  red[threadIdx.x] = s;
  __syncthreads();
  for (int o = 128; o > 0; o >>= 1) {
    if (threadIdx.x < o) red[threadIdx.x] += red[threadIdx.x + o];
    __syncthreads();
  }
  if (threadIdx.x == 0) out[0] = (red[0] + *accum) / (float)NROWS;
}

extern "C" void kernel_launch(void* const* d_in, const int* in_sizes, int n_in,
                              void* d_out, int out_size, void* d_ws, size_t ws_size,
                              hipStream_t stream) {
  (void)in_sizes; (void)n_in; (void)out_size; (void)ws_size;
  const float* ei = (const float*)d_in[0];
  const float* ej = (const float*)d_in[1];

  char* ws = (char*)d_ws;
  _Float16* zhi  = (_Float16*)(ws);                               // 2 MB
  _Float16* zlo  = (_Float16*)(ws + (size_t)2 * 1024 * 1024);     // 2 MB
  float*    den  = (float*)  (ws + (size_t)4 * 1024 * 1024);      // 32 KB
  float*    acc  = (float*)  (ws + (size_t)4 * 1024 * 1024 + NROWS * sizeof(float));

  k_normalize<<<NROWS / 8, 256, 0, stream>>>(ei, ej, zhi, zlo, den, acc);
  k_pos<<<4096 / 8, 256, 0, stream>>>(ei, ej, acc);
  dim3 grid(NROWS / 128, 8);
  k_sim<<<grid, 256, 0, stream>>>(zhi, zlo, den);
  k_final<<<1, 256, 0, stream>>>(den, acc, (float*)d_out);
}
